// WaveNetModel_30932354466402
// MI455X (gfx1250) — matrix operand order repaired and checked by reading, not compiled, not run on hardware
//
#include <hip/hip_runtime.h>
#include <hip/hip_bf16.h>
#include <stdint.h>

typedef __bf16 bf16_t;
typedef __attribute__((ext_vector_type(16))) __bf16 v16bf;
typedef __attribute__((ext_vector_type(8)))  float  v8f;

#define LMAX 9211     // input length (input tensor pitch)
#define HP   9216     // padded pitch for h buffers (16B-aligned rows)
#define TT   64

// ---- helpers -------------------------------------------------------------

__device__ inline bf16_t f2bf(float f) {
  unsigned int u = __float_as_uint(f);
  unsigned int r = u + 0x7FFFu + ((u >> 16) & 1u);   // round-to-nearest-even
  unsigned short s = (unsigned short)(r >> 16);
  return __builtin_bit_cast(bf16_t, s);
}
__device__ inline bf16_t u16bf(unsigned short s) {
  return __builtin_bit_cast(bf16_t, s);
}
// A-fragment (16x32 bf16): lanes 0-15 hold K=0..15, lanes 16-31 hold K=16..31
__device__ inline int aKmap(int e, int lane) {
  return (lane < 16) ? e : e + 16;
}
// B-fragment (32x16 bf16): lanes 0-15 hold K=0..15, lanes 16-31 hold K=16..31
__device__ inline int bKmap(int e, int lane) { return (lane < 16) ? e : e + 16; }

// ---- fast activation -----------------------------------------------------

#if defined(__has_builtin)
#if __has_builtin(__builtin_amdgcn_tanhf)
#define FAST_TANH 1
#endif
#if __has_builtin(__builtin_amdgcn_rcpf)
#define FAST_RCP 1
#endif
#endif

__device__ inline float fast_tanh(float x) {
#ifdef FAST_TANH
  return __builtin_amdgcn_tanhf(x);
#else
  return tanhf(x);
#endif
}
__device__ inline float fast_rcp(float x) {
#ifdef FAST_RCP
  return __builtin_amdgcn_rcpf(x);
#else
  return 1.0f / x;
#endif
}

// ---- async global->LDS staging (CDNA5 ASYNCcnt path), with fallback ------

#if defined(__has_builtin)
#if __has_builtin(__builtin_amdgcn_global_load_async_to_lds_b128)
#define HAVE_ASYNC_COPY 1
#endif
#endif

#ifdef HAVE_ASYNC_COPY
typedef int v4i_ __attribute__((vector_size(16)));
typedef __attribute__((address_space(1))) v4i_ gv4i_t;
typedef __attribute__((address_space(3))) v4i_ lv4i_t;
__device__ inline void async_cp16(void* g, void* l) {
  __builtin_amdgcn_global_load_async_to_lds_b128((gv4i_t*)g, (lv4i_t*)l, 0, 0);
}
__device__ inline void wait_async() {
  asm volatile("s_wait_asynccnt 0" ::: "memory");
}
#endif

// copies `bytes` (multiple of 16) from contiguous global g to LDS l
__device__ inline void stage_bytes(const void* g, void* l, int bytes, int tid) {
#ifdef HAVE_ASYNC_COPY
  int chunks = bytes >> 4;
  for (int t = tid; t < chunks; t += 256)
    async_cp16((char*)g + (size_t)t * 16, (char*)l + (size_t)t * 16);
#else
  int dwords = bytes >> 2;
  const uint32_t* gs = (const uint32_t*)g;
  uint32_t* ld = (uint32_t*)l;
  for (int t = tid; t < dwords; t += 256) ld[t] = gs[t];
#endif
}
__device__ inline void stage_fence() {
#ifdef HAVE_ASYNC_COPY
  wait_async();
#endif
}

// ---- weight packing ------------------------------------------------------

// W1[i] = [[Wf_tap0 Wf_tap1],[Wg_tap0 Wg_tap1]]  (64 x 64) bf16
__global__ void k_pack1(const float* __restrict__ filt, const float* __restrict__ gate,
                        bf16_t* __restrict__ W1) {
  int t = blockIdx.x * 256 + threadIdx.x;
  if (t >= 50 * 64 * 64) return;
  int i = t >> 12;
  int rem = t & 4095;
  int row = rem >> 6, col = rem & 63;
  int c = col & 31, k = col >> 5;
  float v = (row < 32) ? filt[((i * 32 + row) * 32 + c) * 2 + k]
                       : gate[((i * 32 + (row - 32)) * 32 + c) * 2 + k];
  W1[t] = f2bf(v);
}

// WR = res_w bf16 [50][32][32]; WS = skip weights repacked [512][1600]
__global__ void k_pack2(const float* __restrict__ res, const float* __restrict__ skw,
                        bf16_t* __restrict__ WR, bf16_t* __restrict__ WS) {
  int t = blockIdx.x * 256 + threadIdx.x;
  if (t < 51200) {
    WR[t] = f2bf(res[t]);                       // identical layout [i][o][c]
  } else if (t < 51200 + 819200) {
    int u = t - 51200;                          // u = o*1600 + (i*32+c)
    int o = u / 1600;
    int k = u - o * 1600;
    int i = k >> 5, c = k & 31;
    WS[u] = f2bf(skw[((size_t)i * 512 + o) * 32 + c]);
  }
}

// packs end1_w (262144), end2_w (131072), start_w (8192) to bf16
__global__ void k_packE(const float* __restrict__ e1, const float* __restrict__ e2,
                        const float* __restrict__ sw, bf16_t* __restrict__ E1,
                        bf16_t* __restrict__ E2, bf16_t* __restrict__ SW) {
  int t = blockIdx.x * 256 + threadIdx.x;
  if (t < 262144) E1[t] = f2bf(e1[t]);
  else if (t < 393216) E2[t - 262144] = f2bf(e2[t - 262144]);
  else if (t < 401408) SW[t - 393216] = f2bf(sw[t - 393216]);
}

#define XT_PITCH 72   // 64 K values + pad; 144 B, 16B aligned
#define ZT_PITCH 40   // 32 K values + pad;  80 B, 16B aligned

// ---- start 1x1 conv as WMMA GEMM: h(32 x L) = SW(32x256) @ in(256 x L) ---

__global__ __launch_bounds__(256) void k_start(const float* __restrict__ in,
                                               const bf16_t* __restrict__ SW,
                                               float* __restrict__ h) {
  __shared__ __align__(16) bf16_t sXt[64 * ZT_PITCH];
  __shared__ __align__(16) bf16_t sW[32 * 32];
  const int tid = threadIdx.x, wave = tid >> 5, lane = tid & 31;
  const int b = blockIdx.y, t0 = blockIdx.x * 64;
  const float* ib = in + (size_t)b * 256 * LMAX;
  const int mt = wave & 1, nt = wave >> 1;   // 8 waves = 2 M-tiles x 4 N-tiles
  const bool full = (t0 + 64) <= LMAX;
  v8f acc = {};
  for (int k0 = 0; k0 < 256; k0 += 32) {
    if (full) {                                // interior fast path
      for (int t = tid; t < 2048; t += 256) {
        int j = t & 63, r = t >> 6;
        sXt[j * ZT_PITCH + r] = f2bf(ib[(size_t)(k0 + r) * LMAX + t0 + j]);
      }
    } else {
      for (int t = tid; t < 2048; t += 256) {
        int j = t & 63, r = t >> 6;
        int col = t0 + j;
        float v = (col < LMAX) ? ib[(size_t)(k0 + r) * LMAX + col] : 0.0f;
        sXt[j * ZT_PITCH + r] = f2bf(v);
      }
    }
    for (int t = tid; t < 512; t += 256) {     // dword copy of W chunk [32][32]
      int r = t >> 4, cd = t & 15;
      ((uint32_t*)sW)[t] = ((const uint32_t*)(SW + (size_t)r * 256 + k0))[cd];
    }
    __syncthreads();
    v16bf af, bfv;
    int m = lane & 15, n = lane & 15;
    #pragma unroll
    for (int e = 0; e < 16; ++e)
      af[e] = sW[(mt * 16 + m) * 32 + aKmap(e, lane)];
    #pragma unroll
    for (int e = 0; e < 16; ++e)
      bfv[e] = sXt[(nt * 16 + n) * ZT_PITCH + bKmap(e, lane)];
    acc = __builtin_amdgcn_wmma_f32_16x16x32_bf16(false, af, false, bfv,
                                                  (short)0, acc, false, false);
    __syncthreads();
  }
  #pragma unroll
  for (int r = 0; r < 8; ++r) {
    int Mr = mt * 16 + r + ((lane < 16) ? 0 : 8);
    int col = t0 + nt * 16 + (lane & 15);
    if (col < LMAX) h[((size_t)b * 32 + Mr) * HP + col] = acc[r];
  }
}

// ---- fused dilated layer (GEMM1 + activation + residual GEMM) ------------
// z (bf16) for the final 4096-col window is written to Zall[b][layer*32+c][t]
// so the 512-wide skip projection can run later as ONE dense K=1600 GEMM.

__global__ __launch_bounds__(256) void k_layer(
    const float* __restrict__ hsrc, float* __restrict__ hdst,
    const bf16_t* __restrict__ W1, const bf16_t* __restrict__ WR,
    bf16_t* __restrict__ Zall, int layer, int d, int Lin, int Lout, int skipOff) {
  __shared__ __align__(16) char smem[40960];
  bf16_t* sA  = (bf16_t*)smem;              // W1 [64][64]        [0, 8192)
  bf16_t* sXt = (bf16_t*)(smem + 8192);     // X^T [64][72]       [8192, 17408)
  float*  sP  = (float*)(smem + 17408);     // pre [64][64]       [17408, 33792)
  bf16_t* sWr = (bf16_t*)(smem + 33792);    // Wres [32][32]      [33792, 35840)
  bf16_t* sZt = (bf16_t*)(smem + 35840);    // z^T [64][40]       [35840, 40960)
  const int tid = threadIdx.x, wave = tid >> 5, lane = tid & 31;
  const int b = blockIdx.y, j0 = blockIdx.x * TT;
  const float* hb = hsrc + (size_t)b * 32 * HP;

  stage_bytes(W1, sA, 64 * 64 * 2, tid);
  stage_bytes(WR, sWr, 32 * 32 * 2, tid);
  if (j0 + 63 + d < Lin) {                  // interior fast path, no guards
    for (int t = tid; t < 4096; t += 256) {
      int col = t & 63, r = t >> 6, c = r & 31;
      int gcol = j0 + col + ((r >= 32) ? d : 0);
      sXt[col * XT_PITCH + r] = f2bf(hb[c * HP + gcol]);
    }
  } else {
    for (int t = tid; t < 4096; t += 256) {
      int col = t & 63, r = t >> 6, c = r & 31;
      int gcol = j0 + col + ((r >= 32) ? d : 0);
      float v = (gcol < Lin) ? hb[c * HP + gcol] : 0.0f;
      sXt[col * XT_PITCH + r] = f2bf(v);
    }
  }
  stage_fence();
  __syncthreads();

  // GEMM1: pre(64x64) = W1(64x64) @ X(64x64); 16 tiles over 8 waves
  for (int tix = wave * 2; tix < wave * 2 + 2; ++tix) {
    int mt = tix >> 2, nt = tix & 3;
    v8f acc = {};
    #pragma unroll
    for (int kt = 0; kt < 2; ++kt) {
      v16bf af, bfv;
      int m = lane & 15, n = lane & 15;
      #pragma unroll
      for (int e = 0; e < 16; ++e)
        af[e] = sA[(mt * 16 + m) * 64 + kt * 32 + aKmap(e, lane)];
      #pragma unroll
      for (int e = 0; e < 16; ++e)
        bfv[e] = sXt[(nt * 16 + n) * XT_PITCH + kt * 32 + bKmap(e, lane)];
      acc = __builtin_amdgcn_wmma_f32_16x16x32_bf16(false, af, false, bfv,
                                                    (short)0, acc, false, false);
    }
    #pragma unroll
    for (int r = 0; r < 8; ++r) {
      int M = mt * 16 + r + ((lane < 16) ? 0 : 8);
      int N = nt * 16 + (lane & 15);
      sP[M * 64 + N] = acc[r];
    }
  }
  __syncthreads();

  // gated activation; store z^T to LDS and z (final window) to Zall
  bf16_t* Zb = Zall + ((size_t)b * 1600 + layer * 32) * 4096;
  const bool zfull = (j0 >= skipOff) && (j0 + 64 <= Lout);
  if (zfull) {                              // whole tile inside skip window
    for (int t = tid; t < 2048; t += 256) {
      int j = t & 63, c = t >> 6;
      float f = sP[c * 64 + j];
      float g = sP[(c + 32) * 64 + j];
      float z = fast_tanh(f) * fast_rcp(1.0f + __expf(-g));
      bf16_t zb = f2bf(z);
      sZt[j * ZT_PITCH + c] = zb;
      Zb[(size_t)c * 4096 + (j0 + j - skipOff)] = zb;
    }
  } else {
    for (int t = tid; t < 2048; t += 256) {
      int j = t & 63, c = t >> 6;
      float f = sP[c * 64 + j];
      float g = sP[(c + 32) * 64 + j];
      float z = fast_tanh(f) * fast_rcp(1.0f + __expf(-g));
      bf16_t zb = f2bf(z);
      sZt[j * ZT_PITCH + c] = zb;
      int jg = j0 + j;
      int sc = jg - skipOff;
      if (jg < Lout && sc >= 0) Zb[(size_t)c * 4096 + sc] = zb;
    }
  }
  __syncthreads();

  // residual GEMM: h'(32x64) = WR(32x32) @ z(32x64); one WMMA job per wave
  float* hdb = hdst + (size_t)b * 32 * HP;
  {
    int mt = wave & 1, nt = wave >> 1;      // 2 M-tiles x 4 N-tiles
    v16bf af, bfv;
    v8f acc = {};
    int m = lane & 15, n = lane & 15;
    #pragma unroll
    for (int e = 0; e < 16; ++e)
      af[e] = sWr[(mt * 16 + m) * 32 + aKmap(e, lane)];
    #pragma unroll
    for (int e = 0; e < 16; ++e)
      bfv[e] = sZt[(nt * 16 + n) * ZT_PITCH + bKmap(e, lane)];
    acc = __builtin_amdgcn_wmma_f32_16x16x32_bf16(false, af, false, bfv,
                                                  (short)0, acc, false, false);
    #pragma unroll
    for (int r = 0; r < 8; ++r) {
      int M = mt * 16 + r + ((lane < 16) ? 0 : 8);
      int j = j0 + nt * 16 + n;
      if (j < Lout) hdb[M * HP + j] = acc[r] + hb[M * HP + j + d];
    }
  }
}

// ---- skip projection: skp[b](512x4096) = WS(512x1600) @ Zall[b](1600x4096)

__global__ __launch_bounds__(256) void k_gemm_skip(
    const bf16_t* __restrict__ W, const bf16_t* __restrict__ Z,
    float* __restrict__ Y) {
  __shared__ __align__(16) bf16_t sXt[64 * ZT_PITCH];
  __shared__ __align__(16) bf16_t sW[128 * 32];
  const int tid = threadIdx.x, wave = tid >> 5, lane = tid & 31;
  const int b = blockIdx.y, t0 = blockIdx.x * 64, m0 = blockIdx.z * 128;
  const bf16_t* zb = Z + (size_t)b * 1600 * 4096;
  v8f acc[4] = {{}, {}, {}, {}};
  for (int k0 = 0; k0 < 1600; k0 += 32) {
    for (int t = tid; t < 1024; t += 256) {  // 2 bf16 per thread (u32 loads)
      int j2 = (t & 31) * 2, r = t >> 5;
      uint32_t v = *(const uint32_t*)(zb + (size_t)(k0 + r) * 4096 + t0 + j2);
      sXt[j2 * ZT_PITCH + r]       = u16bf((unsigned short)(v & 0xFFFFu));
      sXt[(j2 + 1) * ZT_PITCH + r] = u16bf((unsigned short)(v >> 16));
    }
    for (int t = tid; t < 2048; t += 256) {  // dword copy of W chunk [128][32]
      int r = t >> 4, cd = t & 15;
      ((uint32_t*)sW)[t] =
          ((const uint32_t*)(W + (size_t)(m0 + r) * 1600 + k0))[cd];
    }
    __syncthreads();
    v16bf af;
    int m = lane & 15, n = lane & 15;
    #pragma unroll
    for (int e = 0; e < 16; ++e)
      af[e] = sW[(wave * 16 + m) * 32 + aKmap(e, lane)];
    #pragma unroll
    for (int nt = 0; nt < 4; ++nt) {
      v16bf bfv;
      #pragma unroll
      for (int e = 0; e < 16; ++e)
        bfv[e] = sXt[(nt * 16 + n) * ZT_PITCH + bKmap(e, lane)];
      acc[nt] = __builtin_amdgcn_wmma_f32_16x16x32_bf16(false, af, false, bfv,
                                                        (short)0, acc[nt], false, false);
    }
    __syncthreads();
  }
  #pragma unroll
  for (int nt = 0; nt < 4; ++nt) {
    #pragma unroll
    for (int r = 0; r < 8; ++r) {
      int Mr = m0 + wave * 16 + r + ((lane < 16) ? 0 : 8);
      int col = t0 + nt * 16 + (lane & 15);
      Y[((size_t)b * 512 + Mr) * 4096 + col] = acc[nt][r];
    }
  }
}

// ---- end head: Y[b][m][t] = act( W[M][K] @ act(X[b][:][t]) + bias ) ------

__global__ __launch_bounds__(256) void k_gemm_end(
    const bf16_t* __restrict__ W, const float* __restrict__ X,
    const float* __restrict__ bias, float* __restrict__ Y,
    int M, int Kdim, int reluIn, int reluOut) {
  __shared__ __align__(16) bf16_t sXt[64 * ZT_PITCH];
  __shared__ __align__(16) bf16_t sW[128 * 32];
  const int tid = threadIdx.x, wave = tid >> 5, lane = tid & 31;
  const int b = blockIdx.y, t0 = blockIdx.x * 64, m0 = blockIdx.z * 128;
  const float* xb = X + (size_t)b * Kdim * 4096;
  v8f acc[4] = {{}, {}, {}, {}};
  for (int k0 = 0; k0 < Kdim; k0 += 32) {
    for (int t = tid; t < 2048; t += 256) {
      int j = t & 63, r = t >> 6;
      float v = xb[(size_t)(k0 + r) * 4096 + t0 + j];
      if (reluIn) v = fmaxf(v, 0.0f);
      sXt[j * ZT_PITCH + r] = f2bf(v);
    }
    for (int t = tid; t < 2048; t += 256) {  // dword copy of W chunk [128][32]
      int r = t >> 4, cd = t & 15;
      ((uint32_t*)sW)[t] =
          ((const uint32_t*)(W + (size_t)(m0 + r) * Kdim + k0))[cd];
    }
    __syncthreads();
    v16bf af;
    int m = lane & 15, n = lane & 15;
    #pragma unroll
    for (int e = 0; e < 16; ++e)
      af[e] = sW[(wave * 16 + m) * 32 + aKmap(e, lane)];
    #pragma unroll
    for (int nt = 0; nt < 4; ++nt) {
      v16bf bfv;
      #pragma unroll
      for (int e = 0; e < 16; ++e)
        bfv[e] = sXt[(nt * 16 + n) * ZT_PITCH + bKmap(e, lane)];
      acc[nt] = __builtin_amdgcn_wmma_f32_16x16x32_bf16(false, af, false, bfv,
                                                        (short)0, acc[nt], false, false);
    }
    __syncthreads();
  }
  #pragma unroll
  for (int nt = 0; nt < 4; ++nt) {
    #pragma unroll
    for (int r = 0; r < 8; ++r) {
      int Mr = m0 + wave * 16 + r + ((lane < 16) ? 0 : 8);
      int col = t0 + nt * 16 + (lane & 15);
      float v = acc[nt][r] + bias[Mr];
      if (reluOut) v = fmaxf(v, 0.0f);
      Y[((size_t)b * M + Mr) * 4096 + col] = v;
    }
  }
}

// ---- host orchestration --------------------------------------------------

extern "C" void kernel_launch(void* const* d_in, const int* in_sizes, int n_in,
                              void* d_out, int out_size, void* d_ws, size_t ws_size,
                              hipStream_t stream) {
  (void)in_sizes; (void)n_in; (void)out_size; (void)ws_size;
  const float* input   = (const float*)d_in[0];
  const float* start_w = (const float*)d_in[1];
  const float* filt_w  = (const float*)d_in[2];
  const float* gate_w  = (const float*)d_in[3];
  const float* res_w   = (const float*)d_in[4];
  const float* skip_w  = (const float*)d_in[5];
  const float* e1w     = (const float*)d_in[6];
  const float* e1b     = (const float*)d_in[7];
  const float* e2w     = (const float*)d_in[8];
  const float* e2b     = (const float*)d_in[9];

  char* ws = (char*)d_ws;
  size_t off = 0;
  auto alloc = [&](size_t bytes) {
    void* p = ws + off;
    off = (off + bytes + 255) & ~(size_t)255;
    return p;
  };
  float*  hA   = (float*)alloc(8ull * 32 * HP * 4);
  float*  hB   = (float*)alloc(8ull * 32 * HP * 4);
  float*  skp  = (float*)alloc(8ull * 512 * 4096 * 4);
  bf16_t* Zall = (bf16_t*)alloc(8ull * 1600 * 4096 * 2);
  float*  y1   = (float*)Zall;   // Zall is dead once skp exists; reuse for y1
  bf16_t* W1   = (bf16_t*)alloc(50ull * 64 * 64 * 2);
  bf16_t* WR   = (bf16_t*)alloc(50ull * 32 * 32 * 2);
  bf16_t* WS   = (bf16_t*)alloc(512ull * 1600 * 2);
  bf16_t* E1   = (bf16_t*)alloc(512ull * 512 * 2);
  bf16_t* E2   = (bf16_t*)alloc(256ull * 512 * 2);
  bf16_t* SW   = (bf16_t*)alloc(32ull * 256 * 2);

  k_pack1<<<800, 256, 0, stream>>>(filt_w, gate_w, W1);
  k_pack2<<<3400, 256, 0, stream>>>(res_w, skip_w, WR, WS);
  k_packE<<<1568, 256, 0, stream>>>(e1w, e2w, start_w, E1, E2, SW);
  k_start<<<dim3((LMAX + 63) / 64, 8), 256, 0, stream>>>(input, SW, hA);

  int L = LMAX;
  float* src = hA;
  float* dst = hB;
  for (int i = 0; i < 50; ++i) {
    int d = 1 << (i % 10);
    int Lout = L - d;
    int skipOff = Lout - 4096;
    dim3 grid((Lout + TT - 1) / TT, 8);
    k_layer<<<grid, 256, 0, stream>>>(src, dst, W1 + (size_t)i * 64 * 64,
                                      WR + (size_t)i * 32 * 32, Zall, i, d, L,
                                      Lout, skipOff);
    float* tmp = src; src = dst; dst = tmp;
    L = Lout;
  }

  k_gemm_skip<<<dim3(64, 8, 4), 256, 0, stream>>>(WS, Zall, skp);
  k_gemm_end<<<dim3(64, 8, 4), 256, 0, stream>>>(E1, skp, e1b, y1, 512, 512, 1, 1);
  k_gemm_end<<<dim3(64, 8, 2), 256, 0, stream>>>(E2, y1, e2b, (float*)d_out, 256, 512, 0, 0);
}
